// MultiHeadDotProduct_541165879967
// MI455X (gfx1250) — matrix operand order repaired and checked by reading, not compiled, not run on hardware
//
#include <hip/hip_runtime.h>
#include <hip/hip_bf16.h>
#include <math.h>

// ---------------------------------------------------------------------------
// Multi-head graph attention for MI455X (gfx1250, wave32, WMMA + TDM).
//   Stage 0: convert feats + weights fp32 -> bf16 (once; weights L2-resident)
//   Stage 1: Q/K/V = feats @ W{q,k,v}^T + b   -> fused bf16 WMMA GEMM
//            (A tile staged to LDS by the Tensor Data Mover)
//   Stage 2: per-edge scores + segment max     (atomic f32 max trick)
//   Stage 3: exp(score - max) + segment sum    (atomicAdd)
//   Stage 4: out-accum ACC[c] += alpha * V[r]  (atomicAdd, L2-resident)
//   Stage 5: out = ACC @ Wo^T + bo             -> bf16 WMMA GEMM
// ---------------------------------------------------------------------------

typedef __attribute__((ext_vector_type(16))) __bf16 v16bf;
typedef __attribute__((ext_vector_type(8)))  float  v8f;
typedef __attribute__((ext_vector_type(4)))  unsigned int u32x4;
typedef __attribute__((ext_vector_type(8)))  int  i32x8;
typedef __attribute__((ext_vector_type(4)))  int  i32x4;

#define DIM    128
#define NHEAD  8
#define APITCH 136   // LDS row pitch in ushorts: 272B = 17*16B -> bank-conflict
                     // free b128 column reads, 16B alignment preserved

#if defined(__gfx1250__) && __has_builtin(__builtin_amdgcn_tensor_load_to_lds) && \
    __has_builtin(__builtin_amdgcn_s_wait_tensorcnt)
#define USE_TDM 1
#else
#define USE_TDM 0
#endif

__device__ __forceinline__ unsigned short f2bf_raw(float f) {
  // round-to-nearest-even fp32 -> bf16
  unsigned u = __float_as_uint(f);
  u += 0x7FFFu + ((u >> 16) & 1u);
  return (unsigned short)(u >> 16);
}

// ----------------------- fp32 -> bf16 convert (x4) -------------------------
__global__ __launch_bounds__(256)
void cvt_bf16_kernel(const float* __restrict__ src, unsigned short* __restrict__ dst,
                     long count4)   // count/4, count multiple of 4
{
  const long i = (long)blockIdx.x * blockDim.x + threadIdx.x;
  if (i >= count4) return;
  const float4 f = *(const float4*)(src + i * 4);
  ushort4 o;
  o.x = f2bf_raw(f.x); o.y = f2bf_raw(f.y);
  o.z = f2bf_raw(f.z); o.w = f2bf_raw(f.w);
  *(ushort4*)(dst + i * 4) = o;
}

// ------------------------------ GEMM (WMMA) --------------------------------
// OUT = A @ W^T + bias for NW weight sets sharing the same bf16 A tile.
// 256 threads (8 wave32s). WG handles one 16-row tile of A; wave w owns the
// 16-col output tile [16w,16w+16). Inner loop: pure b128 loads + WMMA.
template <int NW>
__global__ __launch_bounds__(256)
void gemm_wmma_kernel(const unsigned short* __restrict__ Abf, int n,
                      const unsigned short* __restrict__ W0, const float* __restrict__ b0, float* __restrict__ O0,
                      const unsigned short* __restrict__ W1, const float* __restrict__ b1, float* __restrict__ O1,
                      const unsigned short* __restrict__ W2, const float* __restrict__ b2, float* __restrict__ O2)
{
  __shared__ unsigned short sA[16 * APITCH];   // 16x128 bf16 tile, padded pitch

  const int  row0 = blockIdx.x * 16;
  const int  tid  = threadIdx.x;
  const int  wave = tid >> 5;
  const int  lane = tid & 31;
  const bool full = (row0 + 16 <= n);          // uniform: tile fully in range

#if USE_TDM
  // -------- Tensor Data Mover: DMA the 16x128 bf16 tile into LDS ----------
  // 2D descriptor: tile 128 x 16 elements of 2B; rows past n read as zero
  // (OOB vs tensor_dim1). pad_interval=64 DWORDs (one 256B row) and
  // pad_amount=4 DWORDs (16B) reproduce the APITCH=136-ushort LDS pitch.
  if (wave == 0) {
    const unsigned long long ga =
        (unsigned long long)(const void*)(Abf + (size_t)row0 * DIM);
    const unsigned lds_off = (unsigned)(unsigned long long)(const void*)&sA[0];
    const unsigned rem = (row0 < n) ? (unsigned)(n - row0) : 0u;

    u32x4 g0;
    g0[0] = 1u;                                   // count=1, user descriptor
    g0[1] = lds_off;                              // lds_addr
    g0[2] = (unsigned)(ga & 0xFFFFFFFFu);         // global_addr[31:0]
    g0[3] = (unsigned)((ga >> 32) & 0x01FFFFFFu)  // global_addr[56:32]
            | (2u << 30);                         // type = 2 ("image")

    i32x8 g1;
    g1[0] = (int)((1u << 16)    // data_size = 1 (2 bytes)
                | (1u << 20)    // pad_enable
                | (5u << 22)    // pad_interval: 64 DWORDs
                | (3u << 25));  // pad_amount:   4 DWORDs
    g1[1] = (int)(128u << 16);                       // tensor_dim0[15:0]=128
    g1[2] = (int)((rem & 0xFFFFu) << 16);            // tensor_dim1[15:0]
    g1[3] = (int)(((rem >> 16) & 0xFFFFu)            // tensor_dim1[31:16]
                | (128u << 16));                     // tile_dim0 = 128
    g1[4] = 16;                                      // tile_dim1 = 16
    g1[5] = 128;                                     // tensor_dim0_stride=128
    g1[6] = 0;
    g1[7] = 0;

    i32x4 gz4 = {0, 0, 0, 0};
    i32x8 gz8 = {0, 0, 0, 0, 0, 0, 0, 0};
    // 6-arg form (clang-23 / therock-10.0 headers): groups 0..3 + extra + cpol
    __builtin_amdgcn_tensor_load_to_lds(g0, g1, gz4, gz4, gz8, 0);
    __builtin_amdgcn_s_wait_tensorcnt(0);
  }
  __syncthreads();
#else
  // -------- fallback: cooperative staging, 256 chunks of 16B --------------
  {
    const int rr = tid >> 4;            // row 0..15
    const int cc = (tid & 15) * 8;      // ushort offset 0..120
    uint4 val;
    if (full) {
      val = *(const uint4*)(Abf + (size_t)(row0 + rr) * DIM + cc);
    } else {
      val = make_uint4(0u, 0u, 0u, 0u);
      if (row0 + rr < n)
        val = *(const uint4*)(Abf + (size_t)(row0 + rr) * DIM + cc);
    }
    *(uint4*)(sA + rr * APITCH + cc) = val;
  }
  __syncthreads();
#endif

  const int col0   = wave * 16;
  const int m      = lane & 15;           // A row / B column / D column
  const int khalfA = (lane >> 4) << 4;    // A: lanes 16-31 hold K+16 half
  const int khalfB = (lane >> 4) << 4;    // B: lanes 16-31 hold K+16 half

  // Preload all four A fragments (8x ds_load_b128) so the k-loop only has
  // B loads in flight.
  v16bf afrag[DIM / 32];
  {
    const unsigned short* arow = sA + m * APITCH + khalfA;
#pragma unroll
    for (int ks = 0; ks < DIM / 32; ++ks) {
      struct { uint4 lo, hi; } ab;
      ab.lo = *(const uint4*)(arow + ks * 32);
      ab.hi = *(const uint4*)(arow + ks * 32 + 8);
      __builtin_memcpy(&afrag[ks], &ab, 32);
    }
  }

  v8f acc0 = {}; v8f acc1 = {}; v8f acc2 = {};

#pragma unroll
  for (int ks = 0; ks < DIM / 32; ++ks) {
    const int kb = ks * 32;
    // B fragments 32x16 bf16: B[k,nn] = W[col0+nn, k]; 16 contiguous bf16
    const size_t wroff = (size_t)(col0 + m) * DIM + kb + khalfB;
    {
      struct { uint4 lo, hi; } bb;
      bb.lo = *(const uint4*)(W0 + wroff);
      bb.hi = *(const uint4*)(W0 + wroff + 8);
      v16bf bm; __builtin_memcpy(&bm, &bb, 32);
      acc0 = __builtin_amdgcn_wmma_f32_16x16x32_bf16(false, afrag[ks], false, bm,
                                                     (short)0, acc0, false, false);
    }
    if constexpr (NW > 1) {
      struct { uint4 lo, hi; } bb;
      bb.lo = *(const uint4*)(W1 + wroff);
      bb.hi = *(const uint4*)(W1 + wroff + 8);
      v16bf bm; __builtin_memcpy(&bm, &bb, 32);
      acc1 = __builtin_amdgcn_wmma_f32_16x16x32_bf16(false, afrag[ks], false, bm,
                                                     (short)0, acc1, false, false);
    }
    if constexpr (NW > 2) {
      struct { uint4 lo, hi; } bb;
      bb.lo = *(const uint4*)(W2 + wroff);
      bb.hi = *(const uint4*)(W2 + wroff + 8);
      v16bf bm; __builtin_memcpy(&bm, &bb, 32);
      acc2 = __builtin_amdgcn_wmma_f32_16x16x32_bf16(false, afrag[ks], false, bm,
                                                     (short)0, acc2, false, false);
    }
  }

  // Store D tiles (+bias). C/D layout: VGPR i -> M = i + (lane>=16 ? 8:0),
  // N = lane&15. Per row, lanes 0-15 cover 16 consecutive floats (coalesced).
  const int nn = m;
  const int mo = (lane >> 4) << 3;
  {
    const float bias = b0[col0 + nn];
    float* orow = O0 + (size_t)(row0 + mo) * DIM + col0 + nn;
    if (full) {
#pragma unroll
      for (int i = 0; i < 8; ++i) orow[(size_t)i * DIM] = acc0[i] + bias;
    } else {
#pragma unroll
      for (int i = 0; i < 8; ++i)
        if (row0 + mo + i < n) orow[(size_t)i * DIM] = acc0[i] + bias;
    }
  }
  if constexpr (NW > 1) {
    const float bias = b1[col0 + nn];
    float* orow = O1 + (size_t)(row0 + mo) * DIM + col0 + nn;
    if (full) {
#pragma unroll
      for (int i = 0; i < 8; ++i) orow[(size_t)i * DIM] = acc1[i] + bias;
    } else {
#pragma unroll
      for (int i = 0; i < 8; ++i)
        if (row0 + mo + i < n) orow[(size_t)i * DIM] = acc1[i] + bias;
    }
  }
  if constexpr (NW > 2) {
    const float bias = b2[col0 + nn];
    float* orow = O2 + (size_t)(row0 + mo) * DIM + col0 + nn;
    if (full) {
#pragma unroll
      for (int i = 0; i < 8; ++i) orow[(size_t)i * DIM] = acc2[i] + bias;
    } else {
#pragma unroll
      for (int i = 0; i < 8; ++i)
        if (row0 + mo + i < n) orow[(size_t)i * DIM] = acc2[i] + bias;
    }
  }
}

// ------------------------------ init ---------------------------------------
__global__ void init_kernel(float* __restrict__ smax, float* __restrict__ ssum,
                            float* __restrict__ acc, int n)
{
  const size_t i = (size_t)blockIdx.x * blockDim.x + threadIdx.x;
  const size_t nd = (size_t)n * DIM;
  if (i < nd) acc[i] = 0.f;
  const size_t nh = (size_t)n * NHEAD;
  if (i < nh) { smax[i] = -1e30f; ssum[i] = 0.f; }
}

// ---------------------- float atomic max (bit trick) -----------------------
__device__ __forceinline__ void atomicMaxF(float* addr, float val) {
  if (val >= 0.f) atomicMax((int*)addr, __float_as_int(val));
  else            atomicMin((unsigned int*)addr, (unsigned int)__float_as_int(val));
}

// ------------------------- per-edge scores + max ---------------------------
// One wave per edge; lane covers 4 dims; head h = lane/4 (HD=16 = 4 lanes).
__global__ __launch_bounds__(256)
void score_kernel(const int* __restrict__ ei,
                  const float* __restrict__ Q, const float* __restrict__ K,
                  float* __restrict__ scores, float* __restrict__ smax, int ne)
{
  const int warp = (int)(((size_t)blockIdx.x * blockDim.x + threadIdx.x) >> 5);
  const int lane = threadIdx.x & 31;
  if (warp >= ne) return;
  const int2 rc = *(const int2*)(ei + (size_t)warp * 2);
  const int r = rc.x, c = rc.y;
  const float4 q4 = *(const float4*)(Q + (size_t)c * DIM + lane * 4);
  const float4 k4 = *(const float4*)(K + (size_t)r * DIM + lane * 4);
  float p = q4.x * k4.x + q4.y * k4.y + q4.z * k4.z + q4.w * k4.w;
  p += __shfl_xor(p, 1, 32);
  p += __shfl_xor(p, 2, 32);
  if ((lane & 3) == 0) {
    const int h = lane >> 2;
    const float s = p * 0.25f;              // 1/sqrt(HD), HD=16
    scores[(size_t)warp * NHEAD + h] = s;
    atomicMaxF(&smax[(size_t)c * NHEAD + h], s);
  }
}

// ------------------------- exp + segment sum -------------------------------
__global__ __launch_bounds__(256)
void expsum_kernel(const int* __restrict__ ei, float* __restrict__ scores,
                   const float* __restrict__ smax, float* __restrict__ ssum, int ne)
{
  const size_t idx = (size_t)blockIdx.x * blockDim.x + threadIdx.x;
  if (idx >= (size_t)ne * NHEAD) return;
  const int e = (int)(idx >> 3);
  const int h = (int)(idx & 7);
  const int c = ei[e * 2 + 1];
  const float ex = expf(scores[idx] - smax[(size_t)c * NHEAD + h]);
  scores[idx] = ex;
  atomicAdd(&ssum[(size_t)c * NHEAD + h], ex);
}

// --------------------- weighted scatter-accumulate -------------------------
__global__ __launch_bounds__(256)
void accum_kernel(const int* __restrict__ ei, const float* __restrict__ scores,
                  const float* __restrict__ ssum, const float* __restrict__ V,
                  float* __restrict__ acc, int ne)
{
  const int warp = (int)(((size_t)blockIdx.x * blockDim.x + threadIdx.x) >> 5);
  const int lane = threadIdx.x & 31;
  if (warp >= ne) return;
  const int2 rc = *(const int2*)(ei + (size_t)warp * 2);
  const int r = rc.x, c = rc.y;
  const int h = lane >> 2;
  const float ex = scores[(size_t)warp * NHEAD + h];
  const float ss = ssum[(size_t)c * NHEAD + h];
  const float alpha = ex / (ss + 1e-16f);
  const float4 v4 = *(const float4*)(V + (size_t)r * DIM + lane * 4);
  float* dst = acc + (size_t)c * DIM + lane * 4;
  atomicAdd(dst + 0, alpha * v4.x);
  atomicAdd(dst + 1, alpha * v4.y);
  atomicAdd(dst + 2, alpha * v4.z);
  atomicAdd(dst + 3, alpha * v4.w);
}

// ------------------------------ launcher -----------------------------------
extern "C" void kernel_launch(void* const* d_in, const int* in_sizes, int n_in,
                              void* d_out, int out_size, void* d_ws, size_t ws_size,
                              hipStream_t stream)
{
  const float* feats = (const float*)d_in[0];
  const int*   ei    = (const int*)d_in[1];
  // d_in[2] edge_attr: unused by the reference computation
  const float* Wq = (const float*)d_in[3];
  const float* bq = (const float*)d_in[4];
  const float* Wk = (const float*)d_in[5];
  const float* bk = (const float*)d_in[6];
  const float* Wv = (const float*)d_in[7];
  const float* bv = (const float*)d_in[8];
  const float* Wo = (const float*)d_in[9];
  const float* bo = (const float*)d_in[10];
  float* out = (float*)d_out;

  const int n  = in_sizes[0] / DIM;
  const int ne = in_sizes[1] / 2;

  // ---------------- workspace layout ----------------
  // float region
  float* ws   = (float*)d_ws;
  float* Q    = ws;
  float* Kf   = Q    + (size_t)n * DIM;
  float* Vf   = Kf   + (size_t)n * DIM;
  float* ACC  = Vf   + (size_t)n * DIM;
  float* SC   = ACC  + (size_t)n * DIM;
  float* SMAX = SC   + (size_t)ne * NHEAD;
  float* SSUM = SMAX + (size_t)n * NHEAD;
  // bf16 (ushort) region
  unsigned short* FBF   = (unsigned short*)(SSUM + (size_t)n * NHEAD);
  unsigned short* ACCBF = FBF   + (size_t)n * DIM;
  unsigned short* WQBF  = ACCBF + (size_t)n * DIM;
  unsigned short* WKBF  = WQBF  + DIM * DIM;
  unsigned short* WVBF  = WKBF  + DIM * DIM;
  unsigned short* WOBF  = WVBF  + DIM * DIM;

  const int tiles = (n + 15) / 16;

  // 0) fp32 -> bf16 pre-conversion (feats + weights)
  {
    const long c4 = (long)n * DIM / 4;
    cvt_bf16_kernel<<<(unsigned)((c4 + 255) / 256), 256, 0, stream>>>(feats, FBF, c4);
    const long w4 = (long)DIM * DIM / 4;
    const unsigned wb = (unsigned)((w4 + 255) / 256);
    cvt_bf16_kernel<<<wb, 256, 0, stream>>>(Wq, WQBF, w4);
    cvt_bf16_kernel<<<wb, 256, 0, stream>>>(Wk, WKBF, w4);
    cvt_bf16_kernel<<<wb, 256, 0, stream>>>(Wv, WVBF, w4);
    cvt_bf16_kernel<<<wb, 256, 0, stream>>>(Wo, WOBF, w4);
  }

  // 1) fused Q/K/V projection (bf16 WMMA, TDM-staged A tile)
  gemm_wmma_kernel<3><<<tiles, 256, 0, stream>>>(FBF, n,
                                                 WQBF, bq, Q,
                                                 WKBF, bk, Kf,
                                                 WVBF, bv, Vf);
  // 2) init smax/ssum/acc
  {
    const size_t tot = (size_t)n * DIM;
    init_kernel<<<(unsigned)((tot + 255) / 256), 256, 0, stream>>>(SMAX, SSUM, ACC, n);
  }
  // 3) edge scores + segment max (wave per edge)
  score_kernel<<<(unsigned)(((size_t)ne * 32 + 255) / 256), 256, 0, stream>>>(
      ei, Q, Kf, SC, SMAX, ne);
  // 4) exp + segment sum
  expsum_kernel<<<(unsigned)(((size_t)ne * NHEAD + 255) / 256), 256, 0, stream>>>(
      ei, SC, SMAX, SSUM, ne);
  // 5) alpha * V scatter-accumulate
  accum_kernel<<<(unsigned)(((size_t)ne * 32 + 255) / 256), 256, 0, stream>>>(
      ei, SC, SSUM, Vf, ACC, ne);
  // 6) output projection: convert ACC to bf16, then WMMA GEMM
  {
    const long c4 = (long)n * DIM / 4;
    cvt_bf16_kernel<<<(unsigned)((c4 + 255) / 256), 256, 0, stream>>>(ACC, ACCBF, c4);
  }
  gemm_wmma_kernel<1><<<tiles, 256, 0, stream>>>(ACCBF, n,
                                                 WOBF, bo, out,
                                                 nullptr, nullptr, nullptr,
                                                 nullptr, nullptr, nullptr);
}
